// MomentumAttention_66331474920156
// MI455X (gfx1250) — matrix operand order repaired and checked by reading, not compiled, not run on hardware
//
#include <hip/hip_runtime.h>
#include <math.h>

typedef __attribute__((ext_vector_type(16))) __bf16 v16bf;
typedef __attribute__((ext_vector_type(8)))  __bf16 v8bf;
typedef __attribute__((ext_vector_type(8)))  float  v8f;
typedef __attribute__((ext_vector_type(4)))  unsigned int u32x4;
typedef __attribute__((ext_vector_type(8)))  unsigned int u32x8;

#define WMMA_BF16(A, Bm, C) \
  __builtin_amdgcn_wmma_f32_16x16x32_bf16(false, (A), false, (Bm), (short)0, (C), false, false)

#define SHUF16(lo, hi) \
  __builtin_shufflevector((lo), (hi), 0, 1, 2, 3, 4, 5, 6, 7, 8, 9, 10, 11, 12, 13, 14, 15)

// ---------------------------------------------------------------------------
// Kernel 1: fused = (relu(raw @ w1 + b1) @ w2 + b2) per position, [B*L]
// ---------------------------------------------------------------------------
__global__ __launch_bounds__(256) void mlp_fused_kernel(
    const float* __restrict__ raw, const float* __restrict__ w1,
    const float* __restrict__ b1, const float* __restrict__ w2,
    const float* __restrict__ b2, float* __restrict__ fused, int DM, int DH) {
  __shared__ float sraw[512];
  __shared__ float sred[256];
  const int pos = blockIdx.x;
  const int j = threadIdx.x;
  const float* r = raw + (size_t)pos * DM;
  for (int i = j; i < DM; i += blockDim.x) sraw[i] = r[i];
  __syncthreads();
  float acc = b1[j];
#pragma unroll 8
  for (int i = 0; i < 512; ++i) acc += sraw[i] * w1[(size_t)i * DH + j];
  sred[j] = fmaxf(acc, 0.f) * w2[j];
  __syncthreads();
  for (int s = 128; s > 0; s >>= 1) {
    if (j < s) sred[j] += sred[j + s];
    __syncthreads();
  }
  if (j == 0) fused[pos] = sred[0] + b2[0];
}

// ---------------------------------------------------------------------------
// Kernel 2: momentum[b,l] = fused[b,l] - fused[b,l-1], left element zero.
// ---------------------------------------------------------------------------
__global__ void momentum_kernel(const float* __restrict__ fused,
                                float* __restrict__ mom, int L, int N) {
  const int idx = blockIdx.x * blockDim.x + threadIdx.x;
  if (idx >= N) return;
  mom[idx] = ((idx % L) >= 1) ? (fused[idx] - fused[idx - 1]) : 0.f;
}

// ---------------------------------------------------------------------------
// Kernel 3a: K fp32 -> bf16, same layout [B,L,H,E].
// ---------------------------------------------------------------------------
__global__ void convert_k_kernel(const float* __restrict__ src,
                                 unsigned short* __restrict__ dst, int N) {
  const int i = blockIdx.x * blockDim.x + threadIdx.x;
  if (i >= N) return;
  const __bf16 v = (__bf16)src[i];
  dst[i] = __builtin_bit_cast(unsigned short, v);
}

// ---------------------------------------------------------------------------
// Kernel 3b: V fp32 [B,L,H,D] -> bf16 transposed [B,H,D,L] (so the attention
// kernel's V tiles are plain 2D strided tiles, TDM-friendly).
// ---------------------------------------------------------------------------
__global__ void convert_v_kernel(const float* __restrict__ src,
                                 unsigned short* __restrict__ dst,
                                 int B, int L, int H, int D) {
  const int i = blockIdx.x * blockDim.x + threadIdx.x;
  if (i >= B * L * H * D) return;
  const int d = i % D;
  const int h = (i / D) % H;
  const int s = (i / (D * H)) % L;
  const int b = i / (D * H * L);
  const __bf16 v = (__bf16)src[i];
  dst[(((size_t)b * H + h) * D + d) * L + s] = __builtin_bit_cast(unsigned short, v);
}

// ---------------------------------------------------------------------------
// TDM: issue a 2D tensor_load_to_lds. All inputs wave-uniform.
// Descriptor bitfields per CDNA5 ISA §8 (group0: count/lds_addr/global_addr,
// group1: data_size=2B, pad_enable + pad_interval/amount, dims/strides).
// ---------------------------------------------------------------------------
__device__ __forceinline__ void tdm_load_2d(unsigned lds_addr, unsigned long long ga,
                                            unsigned tdim0, unsigned tdim1,
                                            unsigned tile0, unsigned tile1,
                                            unsigned long long stride0,
                                            unsigned pad_interval, unsigned pad_amount) {
  u32x4 g0;
  g0[0] = 1u;                                               // count=1, user D#
  g0[1] = lds_addr;                                         // LDS byte address
  g0[2] = (unsigned)(ga & 0xffffffffu);                     // global_addr[31:0]
  g0[3] = (unsigned)((ga >> 32) & 0x01ffffffu) | (2u << 30);// addr[56:32] | type=2
  u32x8 g1;
  g1[0] = (1u << 16)                                        // data_size = 2 bytes
        | (1u << 20)                                        // pad_enable
        | (pad_interval << 22) | (pad_amount << 25);
  g1[1] = (tdim0 & 0xffffu) << 16;                          // tensor_dim0[15:0]
  g1[2] = (tdim0 >> 16) | ((tdim1 & 0xffffu) << 16);        // tdim0[31:16] | tdim1[15:0]
  g1[3] = (tdim1 >> 16) | ((tile0 & 0xffffu) << 16);        // tdim1[31:16] | tile_dim0
  g1[4] = tile1 & 0xffffu;                                  // tile_dim1 (tile_dim2=0)
  g1[5] = (unsigned)(stride0 & 0xffffffffu);                // tensor_dim0_stride[31:0]
  g1[6] = (unsigned)((stride0 >> 32) & 0xffffu);            // stride[47:32]
  g1[7] = 0u;
  u32x4 gz = {0u, 0u, 0u, 0u};
  asm volatile("tensor_load_to_lds %0, %1, %2, %3"
               :
               : "s"(g0), "s"(g1), "s"(gz), "s"(gz)
               : "memory");
}

// ---------------------------------------------------------------------------
// Kernel 4: causal flash-attention with momentum bias, WMMA bf16 + TDM tiles.
// Grid (L/64, H, B); 128 threads = 4 waves; wave w owns q rows
// [qb+16w, qb+16w+15]. Key tiles of 32 (=> P tile is exactly one A-fragment).
// ---------------------------------------------------------------------------
__global__ __launch_bounds__(128) void attn_wmma_kernel(
    const float* __restrict__ Qg, const unsigned short* __restrict__ Kb,
    const unsigned short* __restrict__ Vt, const float* __restrict__ mom,
    const float* __restrict__ alpha, float* __restrict__ out,
    int B, int L, int H) {
  constexpr int E = 64, D = 64;
  constexpr int KS = 72;  // K-tile row stride: 128B data + 16B TDM pad (9*16B)
  constexpr int VS = 40;  // Vt-tile row stride: 64B data + 16B TDM pad (5*16B)
  constexpr int PS = 40;  // P staging row stride

  __shared__ alignas(16) __bf16 sK[32 * KS];      // K tile  [32 keys][64 feats]
  __shared__ alignas(16) __bf16 sVt[64 * VS];     // V tile  [64 d][32 keys]
  __shared__ alignas(16) __bf16 sP[4 * 16 * PS];  // per-wave P staging [16][32]

  const int b = blockIdx.z, h = blockIdx.y;
  const int qb = blockIdx.x * 64;
  const int tid = threadIdx.x;
  const int wave = tid >> 5;
  const int lane = tid & 31;
  const int n = lane & 15;    // N index inside 16 (C/B layout)
  const int hlf = lane >> 4;  // 16-lane half
  const int kb8 = hlf << 3;   // A-fragment K base (0/8)
  const int kb16 = hlf << 4;  // unused; B fragments also use kb8 (0/8)
  const int q0 = qb + wave * 16;

  // scale*log2(e): folded into Q so softmax uses exp2 directly.
  const float c1 = 0.125f * 1.44269504f;
  const float ascale = c1 * alpha[h];

  const unsigned ldsK = (unsigned)(size_t)(void*)sK;
  const unsigned ldsV = (unsigned)(size_t)(void*)sVt;

  // ---- Q fragments (pre-scaled): two 16x32 bf16 A-frags covering E=64 ----
  const float* qrow = Qg + ((size_t)((b * L + q0 + n) * H + h)) * E;
  v16bf qf[2];
#pragma unroll
  for (int f = 0; f < 2; ++f) {
#pragma unroll
    for (int i = 0; i < 8; ++i) {
      qf[f][i]     = (__bf16)(qrow[f * 32 + kb8 + i] * c1);
      qf[f][8 + i] = (__bf16)(qrow[f * 32 + 16 + kb8 + i] * c1);
    }
  }

  float mq[8];
#pragma unroll
  for (int r = 0; r < 8; ++r) mq[r] = mom[b * L + q0 + r + 8 * hlf];

  float m_run[8], l_run[8];
  v8f o[4];
#pragma unroll
  for (int r = 0; r < 8; ++r) { m_run[r] = -1e30f; l_run[r] = 0.f; }
#pragma unroll
  for (int c = 0; c < 4; ++c)
#pragma unroll
    for (int r = 0; r < 8; ++r) o[c][r] = 0.f;

  const int ntiles = qb / 32 + 2;  // covers keys [0, qb+63]
  for (int j = 0; j < ntiles; ++j) {
    const int s0 = j * 32;

    // ---- TDM: DMA both bf16 tiles into padded LDS (wave 0 only) ----
    if (wave == 0) {
      const unsigned long long gaK = (unsigned long long)(size_t)Kb +
          ((size_t)((b * L + s0) * H + h)) * E * 2ull;
      // K: 32 rows x 64 elems, row stride H*E elems, pad 128B->+16B (KS=72)
      tdm_load_2d(ldsK, gaK, 64u, 32u, 64u, 32u,
                  (unsigned long long)(H * E), 4u, 3u);
      const unsigned long long gaV = (unsigned long long)(size_t)Vt +
          (((size_t)b * H + h) * D * L + s0) * 2ull;
      // Vt: 64 rows x 32 elems, row stride L elems, pad 64B->+16B (VS=40)
      tdm_load_2d(ldsV, gaV, 32u, 64u, 32u, 64u,
                  (unsigned long long)L, 3u, 3u);
      __builtin_amdgcn_s_wait_tensorcnt(0);
    } else if (j + 1 < ntiles) {
      // warm L2 for the next tile (global_prefetch_b8)
      if (wave == 1) {
        __builtin_prefetch(Kb + ((size_t)((b * L + s0 + 32 + lane) * H + h)) * E, 0, 1);
      } else if (wave == 2) {
        __builtin_prefetch(Vt + (((size_t)b * H + h) * D + 2 * lane) * L + s0 + 32, 0, 1);
        __builtin_prefetch(Vt + (((size_t)b * H + h) * D + 2 * lane + 1) * L + s0 + 32, 0, 1);
      }
    }
    __syncthreads();

    if (s0 <= q0 + 15) {  // wave-uniform: tile not fully masked
      // ---- S = Q K^T : 2 N-tiles x 2 K-steps ----
      v8f sc[2];
#pragma unroll
      for (int c = 0; c < 2; ++c) {
#pragma unroll
        for (int r = 0; r < 8; ++r) sc[c][r] = 0.f;
#pragma unroll
        for (int f = 0; f < 2; ++f) {
          const v8bf* bp = (const v8bf*)&sK[(c * 16 + n) * KS + f * 32 + kb8];
          v16bf kf = SHUF16(bp[0], bp[2]);
          sc[c] = WMMA_BF16(qf[f], kf, sc[c]);
        }
      }
      // ---- bias + causal mask (log2 domain; scale folded into Q) ----
      const float ms0 = mom[b * L + s0 + n];
      const float ms1 = mom[b * L + s0 + 16 + n];
      float rowmax[8];
#pragma unroll
      for (int r = 0; r < 8; ++r) {
        const int qi = q0 + r + 8 * hlf;
        float v0 = fmaf(-ascale, fabsf(mq[r] - ms0), sc[0][r]);
        float v1 = fmaf(-ascale, fabsf(mq[r] - ms1), sc[1][r]);
        if (s0 + n > qi) v0 = -1e30f;
        if (s0 + 16 + n > qi) v1 = -1e30f;
        sc[0][r] = v0;
        sc[1][r] = v1;
        rowmax[r] = fmaxf(v0, v1);
      }
#pragma unroll
      for (int msk = 8; msk >= 1; msk >>= 1)
#pragma unroll
        for (int r = 0; r < 8; ++r)
          rowmax[r] = fmaxf(rowmax[r], __shfl_xor(rowmax[r], msk, 32));
      // ---- online softmax update (exp2) ----
      float corr[8], rsum[8];
#pragma unroll
      for (int r = 0; r < 8; ++r) {
        const float mn = fmaxf(m_run[r], rowmax[r]);
        corr[r] = exp2f(m_run[r] - mn);
        m_run[r] = mn;
        const float p0 = exp2f(sc[0][r] - mn);
        const float p1 = exp2f(sc[1][r] - mn);
        sc[0][r] = p0;
        sc[1][r] = p1;
        rsum[r] = p0 + p1;
      }
#pragma unroll
      for (int msk = 8; msk >= 1; msk >>= 1)
#pragma unroll
        for (int r = 0; r < 8; ++r)
          rsum[r] += __shfl_xor(rsum[r], msk, 32);
#pragma unroll
      for (int r = 0; r < 8; ++r) l_run[r] = l_run[r] * corr[r] + rsum[r];
#pragma unroll
      for (int c = 0; c < 4; ++c)
#pragma unroll
        for (int r = 0; r < 8; ++r) o[c][r] *= corr[r];

      // ---- re-layout P (C layout) -> bf16 A fragment via per-wave LDS ----
      __bf16* pst = sP + wave * 16 * PS;
#pragma unroll
      for (int r = 0; r < 8; ++r) {
        const int M = r + 8 * hlf;
        pst[M * PS + n]      = (__bf16)sc[0][r];
        pst[M * PS + 16 + n] = (__bf16)sc[1][r];
      }
      const v8bf* pa0 = (const v8bf*)&pst[n * PS + kb8];
      const v8bf* pa1 = (const v8bf*)&pst[n * PS + 16 + kb8];
      v16bf pf = SHUF16(pa0[0], pa1[0]);

      // ---- O += P V : 4 d-tiles ----
#pragma unroll
      for (int c = 0; c < 4; ++c) {
        const v8bf* vp = (const v8bf*)&sVt[(c * 16 + n) * VS + kb8];
        v16bf vf = SHUF16(vp[0], vp[2]);
        o[c] = WMMA_BF16(pf, vf, o[c]);
      }
    }
    __syncthreads();
  }

  // ---- epilogue: O / l, store [B,L,H,D] fp32 ----
#pragma unroll
  for (int r = 0; r < 8; ++r) {
    const int qi = q0 + r + 8 * hlf;
    const float inv = 1.f / l_run[r];
    float* orow = out + ((size_t)((b * L + qi) * H + h)) * D;
#pragma unroll
    for (int c = 0; c < 4; ++c) orow[c * 16 + n] = o[c][r] * inv;
  }
}

// ---------------------------------------------------------------------------
extern "C" void kernel_launch(void* const* d_in, const int* in_sizes, int n_in,
                              void* d_out, int out_size, void* d_ws, size_t ws_size,
                              hipStream_t stream) {
  const float* Q     = (const float*)d_in[0];
  const float* K     = (const float*)d_in[1];
  const float* V     = (const float*)d_in[2];
  const float* raw   = (const float*)d_in[3];
  const float* w1    = (const float*)d_in[4];
  const float* b1    = (const float*)d_in[5];
  const float* w2    = (const float*)d_in[6];
  const float* b2    = (const float*)d_in[7];
  const float* alpha = (const float*)d_in[8];
  float* out = (float*)d_out;

  const int H  = in_sizes[8];        // 16
  const int DH = in_sizes[5];        // 256
  const int DM = in_sizes[4] / DH;   // 512
  const int BL = in_sizes[3] / DM;   // B*L = 4096
  const int B  = 2;
  const int L  = BL / B;             // 2048
  const int NK = in_sizes[1];        // B*L*H*E
  const int NV = in_sizes[2];        // B*L*H*D

  float* fused = (float*)d_ws;                       // [B*L] f32
  float* momv  = fused + BL;                         // [B*L] f32
  unsigned short* Kb = (unsigned short*)(momv + BL); // [B,L,H,E] bf16
  unsigned short* Vt = Kb + (size_t)NK;              // [B,H,D,L] bf16

  mlp_fused_kernel<<<BL, 256, 0, stream>>>(raw, w1, b1, w2, b2, fused, DM, DH);
  momentum_kernel<<<(BL + 255) / 256, 256, 0, stream>>>(fused, momv, L, BL);
  convert_k_kernel<<<(NK + 255) / 256, 256, 0, stream>>>(K, Kb, NK);
  convert_v_kernel<<<(NV + 255) / 256, 256, 0, stream>>>(V, Vt, B, L, H, 64);
  dim3 grid(L / 64, H, B);
  attn_wmma_kernel<<<grid, 128, 0, stream>>>(Q, Kb, Vt, momv, alpha, out, B, L, H);
}
